// mGRU_420906795588
// MI455X (gfx1250) — matrix operand order repaired and checked by reading, not compiled, not run on hardware
//
#include <hip/hip_runtime.h>

// ---------------------------------------------------------------------------
// mLSTM fused cell for MI455X (gfx1250, wave32, WMMA)
// B=16384, IN=512, UP=1024, HID=128, HN=8, HS=16, KER=4
// 256 threads = 8 wave32 per block, one block per WGP (LDS ~250 KB),
// -> 2 waves per SIMD32 for latency hiding in both WMMA and HBM phases.
// ---------------------------------------------------------------------------

#define BROWS   16384
#define IN_DIM  512
#define UP_DIM  1024
#define HID_DIM 128
#define HN_     8
#define HS_     16
#define ROWS_PB 32          // rows per block (2 WMMA m-tiles)
#define NTHREADS 256        // 8 wave32
#define WAVES   (NTHREADS / 32)

typedef __attribute__((ext_vector_type(16))) _Float16 v16h;
typedef __attribute__((ext_vector_type(8)))  float    v8f;

__device__ __forceinline__ int laneid() { return threadIdx.x & 31; }

// ---- A fragment (16x32 f16, M x K) from row-major f16 in LDS --------------
// layout: lanes 0-15: M=lane, K=k0+0..15
//         lanes 16-31: M=lane-16, K=k0+16..31
__device__ __forceinline__ v16h load_a_lds(const _Float16* Abase, int lda, int k0) {
    const int lane = laneid();
    const _Float16* p = Abase + (lane & 15) * lda + k0 + ((lane & 16) ? 16 : 0);
    v16h a;
#pragma unroll
    for (int t = 0; t < 8; ++t) { a[t] = p[t]; a[t + 8] = p[t + 8]; }
    return a;
}

// ---- B fragment (32x16 f16, K x N) from row-major f16 weight W[N][K] ------
// lanes 0-15: N=lane, K=k0+0..15 ; lanes 16-31: N=lane-16, K=k0+16..31
// => 16 contiguous halves per lane, straight from global (L2-resident).
__device__ __forceinline__ v16h load_b_g(const _Float16* __restrict__ W, int ldw,
                                         int n0, int k0) {
    const int lane = laneid();
    const _Float16* p = W + (size_t)(n0 + (lane & 15)) * ldw + k0 + ((lane & 16) ? 16 : 0);
    v16h b;
#pragma unroll
    for (int t = 0; t < 16; ++t) b[t] = p[t];
    return b;
}

__device__ __forceinline__ v16h load_b_g_guard(const _Float16* __restrict__ W, int ldw,
                                               int nmax, int n0, int k0) {
    const int lane = laneid();
    const int n = n0 + (lane & 15);
    v16h b;
    if (n < nmax) {
        const _Float16* p = W + (size_t)n * ldw + k0 + ((lane & 16) ? 16 : 0);
#pragma unroll
        for (int t = 0; t < 16; ++t) b[t] = p[t];
    } else {
#pragma unroll
        for (int t = 0; t < 16; ++t) b[t] = (_Float16)0.f;
    }
    return b;
}

// ---- one 16x16 output tile: D = A(lds) * W^T(global f16) + bias -----------
__device__ __forceinline__ v8f gemm_tile(const _Float16* Abase, int lda,
                                         const _Float16* __restrict__ W, int ldw,
                                         int K, int n0, const float* __restrict__ bias) {
    const int lane = laneid();
    const float bv = bias ? bias[n0 + (lane & 15)] : 0.f;
    v8f acc;
#pragma unroll
    for (int j = 0; j < 8; ++j) acc[j] = bv;
    for (int k0 = 0; k0 < K; k0 += 32) {
        v16h a = load_a_lds(Abase, lda, k0);
        v16h b = load_b_g(W, ldw, n0, k0);
        acc = __builtin_amdgcn_wmma_f32_16x16x32_f16(false, a, false, b,
                                                     (short)0, acc, false, false);
    }
    return acc;
}

// D layout: element (m,n) -> lane = (n&15) + (m>=8 ? 16:0), vgpr j = m&7
__device__ __forceinline__ void store_d_f32(float* base, int stride, v8f d) {
    const int lane = laneid();
    const int n  = lane & 15;
    const int mo = (lane & 16) ? 8 : 0;
#pragma unroll
    for (int j = 0; j < 8; ++j) base[(mo + j) * stride + n] = d[j];
}

__device__ __forceinline__ void store_d_f16(_Float16* base, int stride, v8f d) {
    const int lane = laneid();
    const int n  = lane & 15;
    const int mo = (lane & 16) ? 8 : 0;
#pragma unroll
    for (int j = 0; j < 8; ++j) base[(mo + j) * stride + n] = (_Float16)d[j];
}

__device__ __forceinline__ float sigm(float x) { return 1.f / (1.f + __expf(-x)); }

// ---------------------------------------------------------------------------
__global__ void cvt_f16(const float* __restrict__ s, _Float16* __restrict__ d, int n) {
    for (int i = blockIdx.x * blockDim.x + threadIdx.x; i < n; i += gridDim.x * blockDim.x)
        d[i] = (_Float16)s[i];
}

// ---------------------------------------------------------------------------
extern "C" __global__ void __launch_bounds__(NTHREADS, 1)
mlstm_fused(const float* __restrict__ seq, const float* __restrict__ c0,
            const float* __restrict__ n0p, const float* __restrict__ m0p,
            const float* __restrict__ ln_w, const float* __restrict__ ln_b,
            const float* __restrict__ b_left, const float* __restrict__ b_right,
            const float* __restrict__ conv_w, const float* __restrict__ conv_b,
            const float* __restrict__ b_i, const float* __restrict__ b_o,
            const float* __restrict__ b_q, const float* __restrict__ b_k,
            const float* __restrict__ b_v, const float* __restrict__ gn_w,
            const float* __restrict__ gn_b, const float* __restrict__ b_down,
            const _Float16* __restrict__ wleft, const _Float16* __restrict__ wright,
            const _Float16* __restrict__ wq, const _Float16* __restrict__ wk,
            const _Float16* __restrict__ wv, const _Float16* __restrict__ wo,
            const _Float16* __restrict__ wskip, const _Float16* __restrict__ wi,
            const _Float16* __restrict__ wdown,
            float* __restrict__ out_y, float* __restrict__ out_c,
            float* __restrict__ out_n, float* __restrict__ out_m)
{
    __shared__ _Float16 s_left[ROWS_PB][UP_DIM];   // 64 KB
    __shared__ _Float16 s_xc[ROWS_PB][UP_DIM];     // 64 KB (x first, then x_conv)
    __shared__ float    s_q[ROWS_PB][HID_DIM];     // 16 KB
    __shared__ float    s_k[ROWS_PB][HID_DIM];     // 16 KB
    __shared__ float    s_v[ROWS_PB][HID_DIM];     // 16 KB
    __shared__ float    s_o[ROWS_PB][HID_DIM];     // 16 KB
    __shared__ float    s_skip[ROWS_PB][HID_DIM];  // 16 KB
    __shared__ float    s_right[ROWS_PB][HID_DIM]; // 16 KB
    __shared__ float    s_h[ROWS_PB][HID_DIM];     // 16 KB
    __shared__ float    s_i[ROWS_PB][HN_];         // 1 KB
    __shared__ _Float16 s_u[ROWS_PB][HID_DIM];     // 8 KB  (~249 KB total < 320 KB/WGP)

    const int tid  = threadIdx.x;
    const int lane = tid & 31;
    const int wave = tid >> 5;
    const int row0 = blockIdx.x * ROWS_PB;

    // ---------------- Phase 1: LayerNorm -> x (f16, LDS) -------------------
    for (int r = wave; r < ROWS_PB; r += WAVES) {
        const float* xr = seq + (size_t)(row0 + r) * IN_DIM;
        float s = 0.f, s2 = 0.f;
        float vals[16];
#pragma unroll
        for (int t = 0; t < 16; ++t) {
            float v = xr[lane + 32 * t];
            vals[t] = v; s += v; s2 += v * v;
        }
#pragma unroll
        for (int off = 16; off; off >>= 1) {
            s  += __shfl_xor(s,  off);
            s2 += __shfl_xor(s2, off);
        }
        const float mu  = s * (1.f / IN_DIM);
        const float var = s2 * (1.f / IN_DIM) - mu * mu;
        const float rs  = rsqrtf(var + 1e-5f);
#pragma unroll
        for (int t = 0; t < 16; ++t) {
            int c = lane + 32 * t;
            s_xc[r][c] = (_Float16)((vals[t] - mu) * rs * ln_w[c] + ln_b[c]);
        }
    }
    __syncthreads();

    // ---------------- Phase 2: left (32x1024) & right (32x128) GEMMs -------
    // 128 left tiles + 16 right tiles, K=512, A = x in s_xc (lda=UP_DIM)
    for (int t = wave; t < 144; t += WAVES) {
        if (t < 128) {
            int mt = t >> 6, nt = t & 63;
            v8f acc = gemm_tile(&s_xc[mt * 16][0], UP_DIM, wleft, IN_DIM, IN_DIM,
                                nt * 16, b_left);
            store_d_f16(&s_left[mt * 16][nt * 16], UP_DIM, acc);
        } else {
            int tt = t - 128, mt = tt >> 3, nt = tt & 7;
            v8f acc = gemm_tile(&s_xc[mt * 16][0], UP_DIM, wright, IN_DIM, IN_DIM,
                                nt * 16, b_right);
            store_d_f32(&s_right[mt * 16][nt * 16], HID_DIM, acc);
        }
    }
    __syncthreads();

    // ---------------- Phase 3: causal conv(4) + SiLU -> x_conv -------------
    {
        const float cw0 = conv_w[0], cw1 = conv_w[1], cw2 = conv_w[2], cw3 = conv_w[3];
        const float cb  = conv_b[0];
        for (int idx = tid; idx < ROWS_PB * UP_DIM; idx += NTHREADS) {
            int r = idx >> 10, c = idx & (UP_DIM - 1);
            float a = cb;
            a += cw3 * (float)s_left[r][c];
            if (c >= 1) a += cw2 * (float)s_left[r][c - 1];
            if (c >= 2) a += cw1 * (float)s_left[r][c - 2];
            if (c >= 3) a += cw0 * (float)s_left[r][c - 3];
            s_xc[r][c] = (_Float16)(a * sigm(a));       // SiLU
        }
    }
    __syncthreads();

    // ---------------- Phase 4: q,k,skip (from x_conv), v,o (from left), i --
    // 5 mats * 16 tiles + 2 i-tiles, K=1024
    for (int t = wave; t < 82; t += WAVES) {
        if (t < 80) {
            int mat = t >> 4, rem = t & 15, mt = rem >> 3, nt = rem & 7;
            int m0 = mt * 16, n0 = nt * 16;
            const _Float16* A; const _Float16* W; const float* bias;
            switch (mat) {
                case 0:  A = &s_xc[m0][0];   W = wq;    bias = b_q;     break;
                case 1:  A = &s_xc[m0][0];   W = wk;    bias = b_k;     break;
                case 2:  A = &s_xc[m0][0];   W = wskip; bias = nullptr; break;
                case 3:  A = &s_left[m0][0]; W = wv;    bias = b_v;     break;
                default: A = &s_left[m0][0]; W = wo;    bias = b_o;     break;
            }
            v8f acc = gemm_tile(A, UP_DIM, W, UP_DIM, UP_DIM, n0, bias);
            switch (mat) {
                case 0: store_d_f32(&s_q[m0][n0], HID_DIM, acc); break;
                case 1: {
#pragma unroll
                    for (int j = 0; j < 8; ++j) acc[j] *= 0.25f;   // 1/sqrt(HS)
                    store_d_f32(&s_k[m0][n0], HID_DIM, acc); break;
                }
                case 2: store_d_f32(&s_skip[m0][n0], HID_DIM, acc); break;
                case 3: store_d_f32(&s_v[m0][n0], HID_DIM, acc); break;
                default: {
#pragma unroll
                    for (int j = 0; j < 8; ++j) acc[j] = sigm(acc[j]); // o gate
                    store_d_f32(&s_o[m0][n0], HID_DIM, acc); break;
                }
            }
        } else {
            // i = x_conv @ w_i.T + b_i   (N=8, padded to a 16-wide tile)
            int m0 = (t - 80) * 16;
            const int n = lane & 15;
            float bv = (n < HN_) ? b_i[n] : 0.f;
            v8f acc;
#pragma unroll
            for (int j = 0; j < 8; ++j) acc[j] = bv;
            for (int k0 = 0; k0 < UP_DIM; k0 += 32) {
                v16h a = load_a_lds(&s_xc[m0][0], UP_DIM, k0);
                v16h b = load_b_g_guard(wi, UP_DIM, HN_, 0, k0);
                acc = __builtin_amdgcn_wmma_f32_16x16x32_f16(false, a, false, b,
                                                             (short)0, acc, false, false);
            }
            if (n < HN_) {
                const int mo = (lane & 16) ? 8 : 0;
#pragma unroll
                for (int j = 0; j < 8; ++j) s_i[m0 + mo + j][n] = acc[j];
            }
        }
    }
    __syncthreads();

    // ---------------- Phase 5: mLSTM state update (streams c0 -> c) --------
    // one (row, head) pair per wave iteration; 256 c-elements over 32 lanes x 8
    for (int pr = wave; pr < ROWS_PB * HN_; pr += WAVES) {
        const int r = pr >> 3, h = pr & 7;
        const int rg = row0 + r;
        const float iv  = s_i[r][h];
        const float fv  = 1.f - iv;
        const float m0v = m0p[(size_t)rg * HN_ + h];
        const float mv  = fmaxf(fv + m0v, iv);
        const float ig  = __expf(iv - mv);
        const float fg  = __expf(fv - mv + m0v);
        if (lane == 0) out_m[(size_t)rg * HN_ + h] = mv;

        // n_new and den = max(n_new . q, 1)
        float den = 0.f;
        if (lane < 16) {
            float kv  = s_k[r][h * HS_ + lane];
            float nv0 = n0p[((size_t)rg * HN_ + h) * HS_ + lane];
            float nn  = fg * nv0 + ig * kv;
            out_n[((size_t)rg * HN_ + h) * HS_ + lane] = nn;
            den = nn * s_q[r][h * HS_ + lane];
        }
#pragma unroll
        for (int off = 8; off; off >>= 1) den += __shfl_xor(den, off);
        den = fmaxf(__shfl(den, 0), 1.f);

        const float* c0b = c0    + ((size_t)rg * HN_ + h) * (HS_ * HS_);
        float*       cb  = out_c + ((size_t)rg * HN_ + h) * (HS_ * HS_);
        if (pr + WAVES < ROWS_PB * HN_)
            __builtin_prefetch(c0b + WAVES * HS_ * HS_, 0, 1);   // next pair's tile

        const int   p_ = lane & 15;
        const float kv = s_k[r][h * HS_ + p_];
        const float qv = s_q[r][h * HS_ + p_];
#pragma unroll
        for (int jj = 0; jj < 8; ++jj) {
            const int e = lane + 32 * jj;          // lanes 0-15: d=2jj, 16-31: d=2jj+1
            const int d = e >> 4;
            float cn = fg * c0b[e] + ig * s_v[r][h * HS_ + d] * kv;
            cb[e] = cn;
            float nump = cn * qv;                  // partial of num[d] = sum_p c[d][p] q[p]
#pragma unroll
            for (int off = 8; off; off >>= 1) nump += __shfl_xor(nump, off);
            if (p_ == 0)
                s_h[r][h * HS_ + d] = s_o[r][h * HS_ + d] * (nump / den);
        }
    }
    __syncthreads();

    // ---------------- Phase 6: GroupNorm + gated combine -> u (f16) --------
    for (int pr = tid; pr < ROWS_PB * HN_; pr += NTHREADS) {
        const int r = pr >> 3, h = pr & 7;
        float hv[HS_], mu = 0.f, s2 = 0.f;
#pragma unroll
        for (int t = 0; t < HS_; ++t) {
            hv[t] = s_h[r][h * HS_ + t];
            mu += hv[t]; s2 += hv[t] * hv[t];
        }
        mu *= (1.f / HS_);
        const float var = s2 * (1.f / HS_) - mu * mu;
        const float rs  = rsqrtf(var + 1e-5f);
#pragma unroll
        for (int t = 0; t < HS_; ++t) {
            const int ch = h * HS_ + t;
            float gn = (hv[t] - mu) * rs * gn_w[ch] + gn_b[ch];
            float rv = s_right[r][ch];
            float u  = (gn + s_skip[r][ch]) * (rv * sigm(rv));
            s_u[r][ch] = (_Float16)u;
        }
    }
    __syncthreads();

    // ---------------- Phase 7: down GEMM (K=128, N=512) + residual ---------
    for (int t = wave; t < 64; t += WAVES) {
        const int mt = t >> 5, nt = t & 31;
        const int m0 = mt * 16, n0 = nt * 16;
        v8f acc = gemm_tile(&s_u[m0][0], HID_DIM, wdown, HID_DIM, HID_DIM, n0, b_down);
        const int n  = lane & 15;
        const int mo = (lane & 16) ? 8 : 0;
#pragma unroll
        for (int j = 0; j < 8; ++j) {
            const size_t rg = (size_t)(row0 + m0 + mo + j);
            const size_t ix = rg * IN_DIM + n0 + n;
            out_y[ix] = acc[j] + seq[ix];
        }
    }
}

// ---------------------------------------------------------------------------
extern "C" void kernel_launch(void* const* d_in, const int* in_sizes, int n_in,
                              void* d_out, int out_size, void* d_ws, size_t ws_size,
                              hipStream_t stream) {
    (void)in_sizes; (void)n_in; (void)out_size; (void)ws_size;
    const float* seq    = (const float*)d_in[0];
    const float* c0     = (const float*)d_in[1];
    const float* n0     = (const float*)d_in[2];
    const float* m0     = (const float*)d_in[3];
    const float* ln_w   = (const float*)d_in[4];
    const float* ln_b   = (const float*)d_in[5];
    const float* w_left = (const float*)d_in[6];
    const float* b_left = (const float*)d_in[7];
    const float* w_right= (const float*)d_in[8];
    const float* b_right= (const float*)d_in[9];
    const float* conv_w = (const float*)d_in[10];
    const float* conv_b = (const float*)d_in[11];
    const float* lskip  = (const float*)d_in[12];
    const float* w_i    = (const float*)d_in[13];
    const float* b_i    = (const float*)d_in[14];
    const float* w_o    = (const float*)d_in[15];
    const float* b_o    = (const float*)d_in[16];
    const float* w_q    = (const float*)d_in[17];
    const float* b_q    = (const float*)d_in[18];
    const float* w_k    = (const float*)d_in[19];
    const float* b_k    = (const float*)d_in[20];
    const float* w_v    = (const float*)d_in[21];
    const float* b_v    = (const float*)d_in[22];
    const float* gn_w   = (const float*)d_in[23];
    const float* gn_b   = (const float*)d_in[24];
    const float* w_down = (const float*)d_in[25];
    const float* b_down = (const float*)d_in[26];

    _Float16* w16 = (_Float16*)d_ws;
    size_t off = 0;
    _Float16* f_wleft  = w16 + off; off += (size_t)UP_DIM  * IN_DIM;   // 1024x512
    _Float16* f_wright = w16 + off; off += (size_t)HID_DIM * IN_DIM;   // 128x512
    _Float16* f_wq     = w16 + off; off += (size_t)HID_DIM * UP_DIM;
    _Float16* f_wk     = w16 + off; off += (size_t)HID_DIM * UP_DIM;
    _Float16* f_wv     = w16 + off; off += (size_t)HID_DIM * UP_DIM;
    _Float16* f_wo     = w16 + off; off += (size_t)HID_DIM * UP_DIM;
    _Float16* f_wskip  = w16 + off; off += (size_t)HID_DIM * UP_DIM;
    _Float16* f_wi     = w16 + off; off += (size_t)HN_     * UP_DIM;
    _Float16* f_wdown  = w16 + off; off += (size_t)IN_DIM  * HID_DIM;  // 512x128

    struct Cvt { const float* s; _Float16* d; int n; };
    const Cvt jobs[9] = {
        { w_left,  f_wleft,  UP_DIM  * IN_DIM  },
        { w_right, f_wright, HID_DIM * IN_DIM  },
        { w_q,     f_wq,     HID_DIM * UP_DIM  },
        { w_k,     f_wk,     HID_DIM * UP_DIM  },
        { w_v,     f_wv,     HID_DIM * UP_DIM  },
        { w_o,     f_wo,     HID_DIM * UP_DIM  },
        { lskip,   f_wskip,  HID_DIM * UP_DIM  },
        { w_i,     f_wi,     HN_     * UP_DIM  },
        { w_down,  f_wdown,  IN_DIM  * HID_DIM },
    };
    for (int j = 0; j < 9; ++j) {
        int blocks = (jobs[j].n + 255) / 256;
        if (blocks > 512) blocks = 512;
        hipLaunchKernelGGL(cvt_f16, dim3(blocks), dim3(256), 0, stream,
                           jobs[j].s, jobs[j].d, jobs[j].n);
    }

    float* out_y = (float*)d_out;
    float* out_c = out_y + (size_t)BROWS * IN_DIM;                 // (B,8,16,16)
    float* out_n = out_c + (size_t)BROWS * HN_ * HS_ * HS_;        // (B,8,16)
    float* out_m = out_n + (size_t)BROWS * HN_ * HS_;              // (B,8)

    hipLaunchKernelGGL(mlstm_fused, dim3(BROWS / ROWS_PB), dim3(NTHREADS), 0, stream,
                       seq, c0, n0, m0, ln_w, ln_b, b_left, b_right, conv_w, conv_b,
                       b_i, b_o, b_q, b_k, b_v, gn_w, gn_b, b_down,
                       f_wleft, f_wright, f_wq, f_wk, f_wv, f_wo, f_wskip, f_wi, f_wdown,
                       out_y, out_c, out_n, out_m);
}
